// CrossAtt_28776280883817
// MI455X (gfx1250) — matrix operand order repaired and checked by reading, not compiled, not run on hardware
//
#include <hip/hip_runtime.h>
#include <cstddef>

typedef __attribute__((ext_vector_type(16))) __bf16 v16bf;
typedef __attribute__((ext_vector_type(8)))  __bf16 v8bf;
typedef __attribute__((ext_vector_type(8)))  float  v8f;

#define GAMMA 0.17677669529663687f   // 32^-0.5

__device__ inline v8bf load8(const __bf16* p) { return *(const v8bf*)p; }

__device__ inline v16bf combine16(v8bf lo, v8bf hi) {
  v16bf r;
#pragma unroll
  for (int i = 0; i < 8; ++i) { r[i] = lo[i]; r[i + 8] = hi[i]; }
  return r;
}

__device__ inline v8f wmma_bf16(v16bf a, v16bf b, v8f c) {
  return __builtin_amdgcn_wmma_f32_16x16x32_bf16(false, a, false, b, (short)0, c,
                                                 false, false);
}

// ---------------------------------------------------------------- stage 0
__global__ void k_f32_to_bf16(const float* __restrict__ in, __bf16* __restrict__ out, int n) {
  int i = blockIdx.x * blockDim.x + threadIdx.x;
  if (i < n) out[i] = (__bf16)in[i];
}

// ---------------------------------------------------------------- stage 1
// in: [B][C][HW] f32  ->  out: [B][HW][C] bf16   grid(HW/32, C/32, B) block(32,8)
__global__ void k_transpose_bf16(const float* __restrict__ in, __bf16* __restrict__ out,
                                 int C, int HW) {
  __shared__ __bf16 t[32][33];
  int b  = blockIdx.z;
  int hw0 = blockIdx.x * 32, c0 = blockIdx.y * 32;
  const float* ip = in + (size_t)b * C * HW;
#pragma unroll
  for (int i = threadIdx.y; i < 32; i += 8)
    t[i][threadIdx.x] = (__bf16)ip[(size_t)(c0 + i) * HW + hw0 + threadIdx.x];
  __syncthreads();
  __bf16* op = out + (size_t)b * HW * C;
#pragma unroll
  for (int i = threadIdx.y; i < 32; i += 8)
    op[(size_t)(hw0 + i) * C + c0 + threadIdx.x] = t[threadIdx.x][i];
}

// ---------------------------------------------------------------- stage 2
// w:[384][Cdim] bf16, xT:[B][HW][Cdim] bf16 -> q,k:[B][4][HW][32], vT:[B][4][32][HW]
// One wave computes a 16(M) x 32(N) tile: A fragment reused across 2 B fragments.
__global__ void k_qkv_gemm(const __bf16* __restrict__ w, const float* __restrict__ bias,
                           const __bf16* __restrict__ xT, int Cdim, int HW,
                           __bf16* __restrict__ q, __bf16* __restrict__ k,
                           __bf16* __restrict__ vT) {
  int lane = threadIdx.x & 31, wid = threadIdx.x >> 5;
  int gw = blockIdx.x * (blockDim.x >> 5) + wid;
  int ntiles = HW >> 5;              // 32-wide N tiles
  int nt = gw % ntiles;
  int t2 = gw / ntiles;
  int mt = t2 % 24;
  int b  = t2 / 24;
  int hf = lane >> 4, lr = lane & 15;
  int m0 = mt * 16, n0 = nt * 32;

  const __bf16* ar  = w  + (size_t)(m0 + lr) * Cdim;
  const __bf16* br0 = xT + ((size_t)b * HW + n0 + lr) * Cdim;
  const __bf16* br1 = xT + ((size_t)b * HW + n0 + 16 + lr) * Cdim;
  v8f acc0 = {}, acc1 = {};
  for (int kk = 0; kk < Cdim; kk += 32) {
    v16bf a   = combine16(load8(ar + kk + 16 * hf), load8(ar + kk + 16 * hf + 8));
    v16bf bb0 = combine16(load8(br0 + kk + 16 * hf), load8(br0 + kk + 16 * hf + 8));
    v16bf bb1 = combine16(load8(br1 + kk + 16 * hf), load8(br1 + kk + 16 * hf + 8));
    acc0 = wmma_bf16(a, bb0, acc0);
    acc1 = wmma_bf16(a, bb1, acc1);
  }
#pragma unroll
  for (int i = 0; i < 8; ++i) {
    int oc = m0 + i + 8 * hf;
    float v0 = acc0[i] + bias[oc];
    float v1 = acc1[i] + bias[oc];
    int t = oc >> 7, h = (oc >> 5) & 3, d = oc & 31;
    int hw0 = n0 + lr, hw1 = n0 + 16 + lr;
    __bf16 b0 = (__bf16)v0, b1 = (__bf16)v1;
    size_t bh = (size_t)(b * 4 + h);
    if (t == 0) {
      q[(bh * HW + hw0) * 32 + d] = b0;
      q[(bh * HW + hw1) * 32 + d] = b1;
    } else if (t == 1) {
      k[(bh * HW + hw0) * 32 + d] = b0;
      k[(bh * HW + hw1) * 32 + d] = b1;
    } else {
      vT[(bh * 32 + d) * HW + hw0] = b0;
      vT[(bh * 32 + d) * HW + hw1] = b1;
    }
  }
}

// ---------------------------------------------------------------- stage 3
// q:[B][4][HWq][32], k:[B][4][HWk][32], vT:[B][4][32][HWk] -> attn:[B][HWq][128]
__global__ void k_attention(const __bf16* __restrict__ q, const __bf16* __restrict__ k,
                            const __bf16* __restrict__ vT, int HWq, int HWk,
                            __bf16* __restrict__ attn_out) {
  __shared__ float  S[8][16][32];
  __shared__ __bf16 P[8][16][32];
  __shared__ float  mrow[8][16], lrow[8][16], rsc[8][16];

  int lane = threadIdx.x & 31, wv = threadIdx.x >> 5;
  int gw = blockIdx.x * 8 + wv;
  int qtiles = HWq >> 4;
  int qt = gw % qtiles;
  int t2 = gw / qtiles;
  int h = t2 & 3, b = t2 >> 2;
  int hf = lane >> 4, lr = lane & 15;

  const __bf16* qp = q  + ((size_t)(b * 4 + h) * HWq) * 32;
  const __bf16* kp = k  + ((size_t)(b * 4 + h) * HWk) * 32;
  const __bf16* vp = vT + ((size_t)(b * 4 + h) * 32) * HWk;

  int q0 = qt * 16;
  const __bf16* qr = qp + (size_t)(q0 + lr) * 32;
  v16bf qf = combine16(load8(qr + 16 * hf), load8(qr + 16 * hf + 8));

  v8f acc0 = {}, acc1 = {};
  if (lane < 16) { mrow[wv][lane] = -3.0e38f; lrow[wv][lane] = 0.f; }
  __syncthreads();

  for (int kb = 0; kb < HWk; kb += 32) {
    const __bf16* kr0 = kp + (size_t)(kb + lr) * 32 + 16 * hf;
    const __bf16* kr1 = kp + (size_t)(kb + 16 + lr) * 32 + 16 * hf;
    v16bf kf0 = combine16(load8(kr0), load8(kr0 + 8));
    v16bf kf1 = combine16(load8(kr1), load8(kr1 + 8));
    if (kb + 32 < HWk) {  // prefetch next K/V chunk -> global_prefetch_b8
      __builtin_prefetch(kp + (size_t)(kb + 32 + lr) * 32, 0, 3);
      __builtin_prefetch(vp + (size_t)lr * HWk + kb + 32, 0, 3);
    }
    v8f z = {};
    v8f s0 = wmma_bf16(qf, kf0, z);
    v8f s1 = wmma_bf16(qf, kf1, z);
#pragma unroll
    for (int i = 0; i < 8; ++i) {
      S[wv][i + 8 * hf][lr]      = s0[i] * GAMMA;
      S[wv][i + 8 * hf][16 + lr] = s1[i] * GAMMA;
    }
    __syncthreads();
    if (lane < 16) {
      int r = lane;
      float mprev = mrow[wv][r];
      float mc = mprev;
#pragma unroll 8
      for (int c = 0; c < 32; ++c) mc = fmaxf(mc, S[wv][r][c]);
      float scale = __expf(mprev - mc);
      float sum = 0.f;
#pragma unroll 8
      for (int c = 0; c < 32; ++c) {
        float p = __expf(S[wv][r][c] - mc);
        sum += p;
        P[wv][r][c] = (__bf16)p;
      }
      lrow[wv][r] = lrow[wv][r] * scale + sum;
      mrow[wv][r] = mc;
      rsc[wv][r]  = scale;
    }
    __syncthreads();
#pragma unroll
    for (int i = 0; i < 8; ++i) {
      float f = rsc[wv][i + 8 * hf];
      acc0[i] *= f; acc1[i] *= f;
    }
    const __bf16* pr = &P[wv][lr][0];
    v16bf pf = combine16(load8(pr + 16 * hf), load8(pr + 16 * hf + 8));
    const __bf16* vr0 = vp + (size_t)lr * HWk + kb + 16 * hf;
    const __bf16* vr1 = vp + (size_t)(16 + lr) * HWk + kb + 16 * hf;
    v16bf vf0 = combine16(load8(vr0), load8(vr0 + 8));
    v16bf vf1 = combine16(load8(vr1), load8(vr1 + 8));
    acc0 = wmma_bf16(pf, vf0, acc0);
    acc1 = wmma_bf16(pf, vf1, acc1);
    __syncthreads();
  }

#pragma unroll
  for (int i = 0; i < 8; ++i) {
    int r = i + 8 * hf;
    float inv = 1.f / lrow[wv][r];
    size_t row = ((size_t)b * HWq + q0 + r) * 128 + h * 32;
    attn_out[row + lr]      = (__bf16)(acc0[i] * inv);
    attn_out[row + 16 + lr] = (__bf16)(acc1[i] * inv);
  }
}

// ---------------------------------------------------------------- stage 4
// wp:[Mdim][128] bf16, attn:[B][HW][128] bf16 -> out:[B][Mdim][HW] f32 (+bias+residual)
// One wave computes a 16(M) x 32(N) tile.
__global__ void k_proj(const __bf16* __restrict__ wp, const float* __restrict__ bias,
                       const __bf16* __restrict__ attn, const float* __restrict__ resid,
                       int Mdim, int HW, float* __restrict__ out) {
  int lane = threadIdx.x & 31, wid = threadIdx.x >> 5;
  int gw = blockIdx.x * (blockDim.x >> 5) + wid;
  int ntiles = HW >> 5, mtiles = Mdim >> 4;
  int nt = gw % ntiles;
  int t2 = gw / ntiles;
  int mt = t2 % mtiles;
  int b  = t2 / mtiles;
  int hf = lane >> 4, lr = lane & 15;
  int m0 = mt * 16, n0 = nt * 32;

  const __bf16* ar  = wp   + (size_t)(m0 + lr) * 128;
  const __bf16* br0 = attn + ((size_t)b * HW + n0 + lr) * 128;
  const __bf16* br1 = attn + ((size_t)b * HW + n0 + 16 + lr) * 128;
  v8f acc0 = {}, acc1 = {};
#pragma unroll
  for (int kk = 0; kk < 128; kk += 32) {
    v16bf a   = combine16(load8(ar + kk + 16 * hf), load8(ar + kk + 16 * hf + 8));
    v16bf bb0 = combine16(load8(br0 + kk + 16 * hf), load8(br0 + kk + 16 * hf + 8));
    v16bf bb1 = combine16(load8(br1 + kk + 16 * hf), load8(br1 + kk + 16 * hf + 8));
    acc0 = wmma_bf16(a, bb0, acc0);
    acc1 = wmma_bf16(a, bb1, acc1);
  }
#pragma unroll
  for (int i = 0; i < 8; ++i) {
    int oc = m0 + i + 8 * hf;
    size_t base = ((size_t)b * Mdim + oc) * HW;
    size_t i0 = base + n0 + lr;
    size_t i1 = base + n0 + 16 + lr;
    out[i0] = acc0[i] + bias[oc] + resid[i0];
    out[i1] = acc1[i] + bias[oc] + resid[i1];
  }
}

// ---------------------------------------------------------------- launch
extern "C" void kernel_launch(void* const* d_in, const int* in_sizes, int n_in,
                              void* d_out, int out_size, void* d_ws, size_t ws_size,
                              hipStream_t stream) {
  (void)in_sizes; (void)n_in; (void)out_size; (void)ws_size;
  const float* x      = (const float*)d_in[0];   // [2,256,64,64]
  const float* y      = (const float*)d_in[1];   // [2,512,32,32]
  const float* wx_qkv = (const float*)d_in[2];   // [384,256]
  const float* bx_qkv = (const float*)d_in[3];   // [384]
  const float* wy_qkv = (const float*)d_in[4];   // [384,512]
  const float* by_qkv = (const float*)d_in[5];   // [384]
  const float* wpx    = (const float*)d_in[6];   // [256,128]
  const float* bpx    = (const float*)d_in[7];   // [256]
  const float* wpy    = (const float*)d_in[8];   // [512,128]
  const float* bpy    = (const float*)d_in[9];   // [512]
  float* out = (float*)d_out;

  const int B = 2, HWX = 4096, HWY = 1024, CX = 256, CY = 512;

  char* ws = (char*)d_ws;
  size_t off = 0;
  auto alloc = [&](size_t bytes) -> void* {
    void* p = ws + off;
    off += (bytes + 255) & ~(size_t)255;
    return p;
  };
  __bf16* wqx_bf = (__bf16*)alloc((size_t)384 * CX * 2);
  __bf16* wqy_bf = (__bf16*)alloc((size_t)384 * CY * 2);
  __bf16* wpx_bf = (__bf16*)alloc((size_t)CX * 128 * 2);
  __bf16* wpy_bf = (__bf16*)alloc((size_t)CY * 128 * 2);
  __bf16* xT     = (__bf16*)alloc((size_t)B * HWX * CX * 2);
  __bf16* yT     = (__bf16*)alloc((size_t)B * HWY * CY * 2);
  __bf16* q_x    = (__bf16*)alloc((size_t)B * 4 * HWX * 32 * 2);
  __bf16* k_x    = (__bf16*)alloc((size_t)B * 4 * HWX * 32 * 2);
  __bf16* vT_x   = (__bf16*)alloc((size_t)B * 4 * 32 * HWX * 2);
  __bf16* q_y    = (__bf16*)alloc((size_t)B * 4 * HWY * 32 * 2);
  __bf16* k_y    = (__bf16*)alloc((size_t)B * 4 * HWY * 32 * 2);
  __bf16* vT_y   = (__bf16*)alloc((size_t)B * 4 * 32 * HWY * 2);
  __bf16* attn_x = (__bf16*)alloc((size_t)B * HWX * 128 * 2);
  __bf16* attn_y = (__bf16*)alloc((size_t)B * HWY * 128 * 2);

  // stage 0: weights -> bf16
  auto cvt = [&](const float* src, __bf16* dst, int n) {
    k_f32_to_bf16<<<(n + 255) / 256, 256, 0, stream>>>(src, dst, n);
  };
  cvt(wx_qkv, wqx_bf, 384 * CX);
  cvt(wy_qkv, wqy_bf, 384 * CY);
  cvt(wpx,    wpx_bf, CX * 128);
  cvt(wpy,    wpy_bf, CY * 128);

  // stage 1: activations -> [B,HW,C] bf16
  k_transpose_bf16<<<dim3(HWX / 32, CX / 32, B), dim3(32, 8), 0, stream>>>(x, xT, CX, HWX);
  k_transpose_bf16<<<dim3(HWY / 32, CY / 32, B), dim3(32, 8), 0, stream>>>(y, yT, CY, HWY);

  // stage 2: QKV GEMMs (one wave per 16x32 tile, 8 waves/block)
  k_qkv_gemm<<<B * 24 * (HWX / 32) / 8, 256, 0, stream>>>(wqx_bf, bx_qkv, xT, CX, HWX,
                                                          q_x, k_x, vT_x);
  k_qkv_gemm<<<B * 24 * (HWY / 32) / 8, 256, 0, stream>>>(wqy_bf, by_qkv, yT, CY, HWY,
                                                          q_y, k_y, vT_y);

  // stage 3: cross attention (x attends to y, y attends to x)
  k_attention<<<B * 4 * (HWX / 16) / 8, 256, 0, stream>>>(q_x, k_y, vT_y, HWX, HWY, attn_x);
  k_attention<<<B * 4 * (HWY / 16) / 8, 256, 0, stream>>>(q_y, k_x, vT_x, HWY, HWX, attn_y);

  // stage 4: output projection + bias + residual
  k_proj<<<B * (CX / 16) * (HWX / 32) / 8, 256, 0, stream>>>(wpx_bf, bpx, attn_x, x,
                                                             CX, HWX, out);
  k_proj<<<B * (CY / 16) * (HWY / 32) / 8, 256, 0, stream>>>(wpy_bf, bpy, attn_y, y,
                                                             CY, HWY,
                                                             out + (size_t)B * CX * HWX);
}
